// DensityEstimator_48541720379658
// MI455X (gfx1250) — matrix operand order repaired and checked by reading, not compiled, not run on hardware
//
#include <hip/hip_runtime.h>
#include <hip/hip_bf16.h>

// DensityEstimator on MI455X (gfx1250, wave32):
//   d2 = sq[i] + sq[j] - 2*A@A^T  (A: 8192x1024 fp32), per-row 6th-smallest,
//   density = 1/(sqrt(d2)+1e-8).
// Strategy:
//   1) split_norm_kernel: fp32 -> bf16 hi/lo matrices (in d_ws) + row norms.
//   2) density_kernel: fused tiled GEMM via v_wmma_f32_16x16x32_bf16
//      (3 WMMAs per K=32 chunk: hh + he + eh ~ fp32 accuracy) with in-register
//      per-column top-6 selection (symmetry: row-kth == column-kth).
// The 2x16MB bf16 matrices are L2-resident (192MB L2), so panel re-reads are
// cheap; HBM sees ~32MB in + 32KB out.

typedef __bf16 bf16_t;
typedef __attribute__((ext_vector_type(8)))  __bf16 v8bf;
typedef __attribute__((ext_vector_type(16))) __bf16 v16bf;
typedef __attribute__((ext_vector_type(8)))  float  v8f;

#define N_PTS 8192
#define K_DIM 1024

__device__ __forceinline__ v16bf cat16(v8bf a, v8bf b) {
  return __builtin_shufflevector(a, b, 0, 1, 2, 3, 4, 5, 6, 7,
                                       8, 9, 10, 11, 12, 13, 14, 15);
}

// Branchless insert into ascending sorted 6-element list (keeps 6 smallest).
__device__ __forceinline__ void top6_insert(float v, float t[6]) {
  float x = v;
#pragma unroll
  for (int i = 0; i < 6; ++i) {
    float lo = fminf(t[i], x);
    float hi = fmaxf(t[i], x);
    t[i] = lo;
    x = hi;
  }
}

// ---------------------------------------------------------------------------
// Kernel 1: per-row fp32 -> bf16 hi/lo split + squared-norm reduction.
// One 256-thread block per row.
// ---------------------------------------------------------------------------
__global__ __launch_bounds__(256)
void DensityEstimator_split_norm_kernel(const float* __restrict__ F,
                                        bf16_t* __restrict__ Ah,
                                        bf16_t* __restrict__ Al,
                                        float* __restrict__ sq) {
  const int row = blockIdx.x;
  const float* fr = F + (size_t)row * K_DIM;
  bf16_t* ah = Ah + (size_t)row * K_DIM;
  bf16_t* al = Al + (size_t)row * K_DIM;

  float s = 0.0f;
#pragma unroll
  for (int j = threadIdx.x; j < K_DIM; j += 256) {
    float x = fr[j];
    __bf16 h = (__bf16)x;            // RNE truncate to bf16
    __bf16 l = (__bf16)(x - (float)h);  // residual
    ah[j] = h;
    al[j] = l;
    s += x * x;
  }
  // wave32 reduction
#pragma unroll
  for (int off = 16; off > 0; off >>= 1) s += __shfl_down(s, off, 32);
  __shared__ float ls[8];
  const int wid = threadIdx.x >> 5;
  const int lane = threadIdx.x & 31;
  if (lane == 0) ls[wid] = s;
  __syncthreads();
  if (threadIdx.x == 0) {
    float t = 0.0f;
#pragma unroll
    for (int i = 0; i < 8; ++i) t += ls[i];
    sq[row] = t;
  }
}

// ---------------------------------------------------------------------------
// Kernel 2: fused distance GEMM + per-column top-6 + density.
// Grid: 256 blocks x 256 threads (8 waves). Each block owns 32 output
// columns (2 x 16-col WMMA tiles per wave); waves stride over the 512
// 16-row tiles (512 % 8 == 0 -> uniform trip count, EXEC all-ones at WMMA).
// ---------------------------------------------------------------------------
__global__ __launch_bounds__(256)
void DensityEstimator_density_kernel(const bf16_t* __restrict__ Ah,
                                     const bf16_t* __restrict__ Al,
                                     const float* __restrict__ sq,
                                     float* __restrict__ out) {
  const int lane = threadIdx.x & 31;
  const int wave = threadIdx.x >> 5;
  const int colBlk = blockIdx.x;                 // 32 columns per block
  const int nLow = lane & 15;
  const int n0 = colBlk * 32 + nLow;             // column tile 0
  const int n1 = n0 + 16;                        // column tile 1

  // Documented gfx1250 16-bit WMMA fragment layouts:
  //  A (16x32, MxK): lane half selects K sub-offset 0/8; VGPR0-3 = K[kA..kA+7],
  //                  VGPR4-7 = K[kA+16..kA+23]  -> two contiguous b128 loads.
  //  B (32x16, KxN): lane half selects K sub-offset 0/8; VGPR0-3 = K[kB..kB+7],
  //                  VGPR4-7 = K[kB+16..kB+23]  -> two contiguous b128 loads.
  const int kA = (lane < 16) ? 0 : 8;
  const int kB = (lane < 16) ? 0 : 8;
  const int mOff = (lane < 16) ? 0 : 8;          // C: VGPR r -> M = r + mOff

  const bf16_t* bh0 = Ah + (size_t)n0 * K_DIM;
  const bf16_t* bl0 = Al + (size_t)n0 * K_DIM;
  const bf16_t* bh1 = Ah + (size_t)n1 * K_DIM;
  const bf16_t* bl1 = Al + (size_t)n1 * K_DIM;

  const float INF = __builtin_inff();
  float t0[6], t1[6];
#pragma unroll
  for (int i = 0; i < 6; ++i) { t0[i] = INF; t1[i] = INF; }

  for (int rt = wave; rt < N_PTS / 16; rt += 8) {
    const int rowBase = rt * 16;
    const bf16_t* arh = Ah + (size_t)(rowBase + nLow) * K_DIM;
    const bf16_t* arl = Al + (size_t)(rowBase + nLow) * K_DIM;

    v8f acc0 = {0.f, 0.f, 0.f, 0.f, 0.f, 0.f, 0.f, 0.f};
    v8f acc1 = {0.f, 0.f, 0.f, 0.f, 0.f, 0.f, 0.f, 0.f};

#pragma unroll 4
    for (int k = 0; k < K_DIM; k += 32) {
      // A fragments (shared across both column tiles)
      v16bf aH = cat16(*(const v8bf*)(arh + k + kA),
                       *(const v8bf*)(arh + k + kA + 16));
      v16bf aL = cat16(*(const v8bf*)(arl + k + kA),
                       *(const v8bf*)(arl + k + kA + 16));
      // B fragments, column tile 0 and 1
      v16bf b0H = cat16(*(const v8bf*)(bh0 + k + kB),
                        *(const v8bf*)(bh0 + k + kB + 16));
      v16bf b0L = cat16(*(const v8bf*)(bl0 + k + kB),
                        *(const v8bf*)(bl0 + k + kB + 16));
      v16bf b1H = cat16(*(const v8bf*)(bh1 + k + kB),
                        *(const v8bf*)(bh1 + k + kB + 16));
      v16bf b1L = cat16(*(const v8bf*)(bl1 + k + kB),
                        *(const v8bf*)(bl1 + k + kB + 16));

      // split-fp32 product: x.y ~ hh + he + eh (fp32 accumulate)
      acc0 = __builtin_amdgcn_wmma_f32_16x16x32_bf16(false, aH, false, b0H,
                                                     (short)0, acc0, false, false);
      acc0 = __builtin_amdgcn_wmma_f32_16x16x32_bf16(false, aH, false, b0L,
                                                     (short)0, acc0, false, false);
      acc0 = __builtin_amdgcn_wmma_f32_16x16x32_bf16(false, aL, false, b0H,
                                                     (short)0, acc0, false, false);
      acc1 = __builtin_amdgcn_wmma_f32_16x16x32_bf16(false, aH, false, b1H,
                                                     (short)0, acc1, false, false);
      acc1 = __builtin_amdgcn_wmma_f32_16x16x32_bf16(false, aH, false, b1L,
                                                     (short)0, acc1, false, false);
      acc1 = __builtin_amdgcn_wmma_f32_16x16x32_bf16(false, aL, false, b1H,
                                                     (short)0, acc1, false, false);
    }

    // partial distances: sq[m] - 2*dot  (column norm added once at the end;
    // it's constant per column so the ranking is unaffected)
    float4 s0 = *(const float4*)(sq + rowBase + mOff);
    float4 s1 = *(const float4*)(sq + rowBase + mOff + 4);
    float sr[8] = {s0.x, s0.y, s0.z, s0.w, s1.x, s1.y, s1.z, s1.w};
#pragma unroll
    for (int r = 0; r < 8; ++r) {
      float d0 = sr[r] - 2.0f * acc0[r];
      float d1 = sr[r] - 2.0f * acc1[r];
      float p0 = __shfl_xor(d0, 16, 32);   // partner lane holds rows m+/-8
      float p1 = __shfl_xor(d1, 16, 32);
      top6_insert(d0, t0);
      top6_insert(p0, t0);
      top6_insert(d1, t1);
      top6_insert(p1, t1);
    }
  }

  // cross-wave merge: lanes 0-15 hold the canonical per-column top-6
  __shared__ float sm[8][32][6];
  if (lane < 16) {
#pragma unroll
    for (int i = 0; i < 6; ++i) {
      sm[wave][nLow][i] = t0[i];
      sm[wave][nLow + 16][i] = t1[i];
    }
  }
  __syncthreads();

  if (threadIdx.x < 32) {
    const int n = colBlk * 32 + threadIdx.x;
    float t[6];
#pragma unroll
    for (int i = 0; i < 6; ++i) t[i] = INF;
#pragma unroll
    for (int w = 0; w < 8; ++w)
#pragma unroll
      for (int i = 0; i < 6; ++i) top6_insert(sm[w][threadIdx.x][i], t);

    // t[5] = 6th smallest (index K_NEIGHBORS=5, self-distance 0 included)
    float d2 = fmaxf(t[5] + sq[n], 0.0f);
    float dist = sqrtf(d2);
    out[n] = 1.0f / (dist + 1e-8f);
  }
}

// ---------------------------------------------------------------------------
extern "C" void kernel_launch(void* const* d_in, const int* in_sizes, int n_in,
                              void* d_out, int out_size, void* d_ws, size_t ws_size,
                              hipStream_t stream) {
  (void)in_sizes; (void)n_in; (void)out_size; (void)ws_size;
  const float* features = (const float*)d_in[0];   // [8192,1024] fp32
  float* out = (float*)d_out;                      // [8192] fp32

  // workspace: Ah (16MB bf16) | Al (16MB bf16) | sq (32KB fp32)  ~= 33.6MB
  bf16_t* Ah = (bf16_t*)d_ws;
  bf16_t* Al = Ah + (size_t)N_PTS * K_DIM;
  float*  sq = (float*)(Al + (size_t)N_PTS * K_DIM);

  DensityEstimator_split_norm_kernel<<<N_PTS, 256, 0, stream>>>(features, Ah, Al, sq);
  DensityEstimator_density_kernel<<<N_PTS / 32, 256, 0, stream>>>(Ah, Al, sq, out);
}
